// Net_NNCONV_45904610459946
// MI455X (gfx1250) — matrix operand order repaired and checked by reading, not compiled, not run on hardware
//
#include <hip/hip_runtime.h>
#include <hip/hip_bf16.h>
#include <math.h>

// ---------------- problem constants (match reference) ----------------
#define DIM       64
#define WIDTH     128
#define IN_DIM    16
#define NN        50000
#define NE        100000
#define NG        2500
#define NSTEPS    5
#define DD        4096   // DIM*DIM

typedef __attribute__((ext_vector_type(16))) __bf16          v16bf;
typedef __attribute__((ext_vector_type(16))) unsigned short  v16us;
typedef __attribute__((ext_vector_type(8)))  float           v8f;

// ---------------- helpers ----------------
__device__ __forceinline__ unsigned short f2bf(float f) {
    union { float f; unsigned u; } v; v.f = f;
    unsigned u = v.u;
    unsigned r = u + 0x7FFFu + ((u >> 16) & 1u);   // round-to-nearest-even
    return (unsigned short)(r >> 16);
}
__device__ __forceinline__ float bf2f(unsigned short h) {
    union { unsigned u; float f; } v; v.u = ((unsigned)h) << 16;
    return v.f;
}
__device__ __forceinline__ float sigmoidf(float x) { return 1.0f / (1.0f + expf(-x)); }

// order-preserving float<->uint encoding for atomic segment-max
__device__ __forceinline__ unsigned f2ord(float f) {
    union { float f; unsigned u; } v; v.f = f;
    return (v.u & 0x80000000u) ? ~v.u : (v.u | 0x80000000u);
}
__device__ __forceinline__ float ord2f(unsigned e) {
    union { unsigned u; float f; } v;
    v.u = (e & 0x80000000u) ? (e & 0x7FFFFFFFu) : ~e;
    return v.f;
}

__device__ __forceinline__ v8f wmma_bf16(v16us a, v16us b, v8f c) {
    return __builtin_amdgcn_wmma_f32_16x16x32_bf16(
        false, __builtin_bit_cast(v16bf, a),
        false, __builtin_bit_cast(v16bf, b),
        (short)0, c, false, false);
}

// A-fragment (16x32 bf16) position for element (row m, k-local kl):
//   lane = (hi<<4)|m, slot s; hi/s from the ISA half-wave K pattern.
__device__ __forceinline__ void a_slot(int kl, int& hi, int& s) {
    int q = kl >> 3;                 // 0..3
    hi = q & 1;
    s  = ((q >> 1) << 3) + (kl & 7);
}

// ---------------- generic utility kernels ----------------
__global__ __launch_bounds__(256) void k_zero_f32(float* p, int n) {
    int i = blockIdx.x * blockDim.x + threadIdx.x;
    if (i < n) p[i] = 0.0f;
}

// pack a [K][N] f32 matrix (optionally given as [N][K], trans=1) into WMMA
// B-fragment-major bf16: dst[((kt*NT+nt)*32 + lane)*16 + slot],
// lane: n = nt*16 + (lane&15); k = kt*32 + (slot>>3)*16 + (lane>>4)*8 + (slot&7)
__global__ __launch_bounds__(256) void k_pack_bfrag(const float* __restrict__ src,
                                                    unsigned short* __restrict__ dst,
                                                    int K, int N, int trans) {
    int i = blockIdx.x * blockDim.x + threadIdx.x;
    if (i >= K * N) return;
    int slot = i & 15;
    int lane = (i >> 4) & 31;
    int f    = i >> 9;
    int NT = N >> 4;
    int kt = f / NT, nt = f % NT;
    int n = nt * 16 + (lane & 15);
    int k = kt * 32 + ((slot >> 3) << 4) + ((lane >> 4) << 3) + (slot & 7);
    float v = trans ? src[n * K + k] : src[k * N + n];
    dst[i] = f2bf(v);
}

// ---------------- lin0: h = relu([x|z] @ W0 + b0); writes f32 h + bf16 A-frags ----------------
__global__ __launch_bounds__(256) void k_lin0(const float* __restrict__ x, const float* __restrict__ z,
                                              const float* __restrict__ w, const float* __restrict__ b,
                                              float* __restrict__ h, unsigned short* __restrict__ hfrag) {
    int i = blockIdx.x * blockDim.x + threadIdx.x;
    if (i >= NN * DIM) return;
    int n = i / DIM, o = i % DIM;
    float acc = b[o];
    #pragma unroll
    for (int k = 0; k < IN_DIM - 1; ++k) acc += x[n * (IN_DIM - 1) + k] * w[k * DIM + o];
    acc += z[n] * w[(IN_DIM - 1) * DIM + o];
    acc = fmaxf(acc, 0.0f);
    h[i] = acc;
    int mt = n >> 4, m = n & 15;
    int kt = o >> 5, kl = o & 31, hi, s;
    a_slot(kl, hi, s);
    hfrag[(((size_t)mt * 2 + kt) * 32 + ((hi << 4) | m)) * 16 + s] = f2bf(acc);
}

// ---------------- in-degree ----------------
__global__ __launch_bounds__(256) void k_deg(const int* __restrict__ dst, float* __restrict__ denom) {
    int e = blockIdx.x * blockDim.x + threadIdx.x;
    if (e < NE) atomicAdd(&denom[dst[e]], 1.0f);
}
__global__ __launch_bounds__(256) void k_deg_fin(float* denom) {
    int n = blockIdx.x * blockDim.x + threadIdx.x;
    if (n < NN) denom[n] = fmaxf(denom[n], 1.0f);
}

// ---------------- edge MLP layer 1: hidden = relu(attr @ W1 + b1) -> A-frags ----------------
__global__ __launch_bounds__(256) void k_edge_hidden(const float* __restrict__ attr,
                                                     const float* __restrict__ w1, const float* __restrict__ b1,
                                                     unsigned short* __restrict__ afrag) {
    int i = blockIdx.x * blockDim.x + threadIdx.x;
    if (i >= NE * WIDTH) return;
    int e = i / WIDTH, k = i % WIDTH;
    float acc = b1[k];
    #pragma unroll
    for (int j = 0; j < 5; ++j) acc += attr[e * 5 + j] * w1[j * WIDTH + k];
    int mt = e >> 4, m = e & 15;
    int kt = k >> 5, kl = k & 31, hi, s;
    a_slot(kl, hi, s);
    afrag[(((size_t)mt * 4 + kt) * 32 + ((hi << 4) | m)) * 16 + s] = f2bf(fmaxf(acc, 0.0f));
}

// ---------------- edge MLP layer 2 (WMMA): ew^T = (hidden @ W2 + b2), per-edge transposed ----
// One wave: 2 M-tiles (32 edges) x 64 cols; A/B via b128 fragment loads; 32 wmma.
__global__ __launch_bounds__(128) void k_edge_ew(const unsigned short* __restrict__ afrag,
                                                 const unsigned short* __restrict__ bfrag,
                                                 const float* __restrict__ b2,
                                                 unsigned short* __restrict__ ewt) {
    int wave = blockIdx.x * (blockDim.x >> 5) + (threadIdx.x >> 5);
    const int NGRP = 64;
    int total = (NE / 32) * NGRP;
    if (wave >= total) return;                    // wave-uniform
    int mt2 = wave / NGRP;                        // pair of M-tiles
    int ng  = wave % NGRP;                        // 64-column group
    int lane = threadIdx.x & 31;
    int lp = lane & 15, hi = lane >> 4;

    v8f acc[2][4] = {};
    #pragma unroll
    for (int kt = 0; kt < 4; ++kt) {
        v16us a0 = *(const v16us*)(afrag + ((((size_t)mt2 * 2 + 0) * 4 + kt) * 32 + lane) * 16);
        v16us a1 = *(const v16us*)(afrag + ((((size_t)mt2 * 2 + 1) * 4 + kt) * 32 + lane) * 16);
        #pragma unroll
        for (int j = 0; j < 4; ++j) {
            int nt = ng * 4 + j;
            v16us b = *(const v16us*)(bfrag + (((size_t)kt * 256 + nt) * 32 + lane) * 16);
            acc[0][j] = wmma_bf16(a0, b, acc[0][j]);
            acc[1][j] = wmma_bf16(a1, b, acc[1][j]);
        }
    }
    #pragma unroll
    for (int i2 = 0; i2 < 2; ++i2) {
        #pragma unroll
        for (int j = 0; j < 4; ++j) {
            int col = ng * 64 + j * 16 + lp;          // flattened (i,o): col = i*64+o
            int io  = (col & 63) * 64 + (col >> 6);   // per-edge transposed index [o][i]
            float bias = b2[col];
            #pragma unroll
            for (int r = 0; r < 8; ++r) {
                int e = (mt2 * 2 + i2) * 16 + hi * 8 + r;
                ewt[(size_t)e * DD + io] = f2bf(acc[i2][j][r] + bias);
            }
        }
    }
}

// ---------------- message: msg_e = h[src_e] @ ew_e ; scatter-add -> aggr ----------------
// ew stored per-edge [o][i] bf16 -> each thread streams 128B with 8 b128 loads.
__global__ __launch_bounds__(64) void k_msg(const float* __restrict__ h,
                                            const unsigned short* __restrict__ ewt,
                                            const int* __restrict__ src, const int* __restrict__ dst,
                                            float* __restrict__ aggr) {
    int e = blockIdx.x;
    int o = threadIdx.x;
    __shared__ float hs[DIM];
    hs[o] = h[(size_t)src[e] * DIM + o];
    __syncthreads();
    const unsigned short* w = ewt + (size_t)e * DD + o * DIM;
    float acc = 0.0f;
    #pragma unroll
    for (int c = 0; c < 8; ++c) {
        uint4 d = *(const uint4*)(w + c * 8);
        unsigned dd[4] = { d.x, d.y, d.z, d.w };
        #pragma unroll
        for (int q = 0; q < 4; ++q) {
            acc += hs[c * 8 + 2 * q]     * bf2f((unsigned short)(dd[q] & 0xFFFFu));
            acc += hs[c * 8 + 2 * q + 1] * bf2f((unsigned short)(dd[q] >> 16));
        }
    }
    atomicAdd(&aggr[(size_t)dst[e] * DIM + o], acc);
}

// ---------------- fused node update (WMMA): NNConv root + GRU ----------------
// One wave per 16-node tile; all operands as pre-swizzled fragments (b128 loads).
__global__ __launch_bounds__(32) void k_node(float* __restrict__ h,
                                             unsigned short* __restrict__ hfrag,
                                             const float* __restrict__ aggr,
                                             const float* __restrict__ denom,
                                             const unsigned short* __restrict__ rootf,  // 2x4 frags
                                             const float* __restrict__ cbias,
                                             const unsigned short* __restrict__ witf,   // 2x12 frags
                                             const unsigned short* __restrict__ whtf,   // 2x12 frags
                                             const float* __restrict__ gbi,
                                             const float* __restrict__ gbh) {
    int lane = threadIdx.x;
    int lp = lane & 15, hi = lane >> 4;
    int tile = blockIdx.x;
    int row0 = tile * 16;

    __shared__ unsigned short mfrag[2 * 32 * 16];
    __shared__ float gi_s[16 * 192];
    __shared__ float gh_s[16 * 192];

    // A fragments of h (bf16 mirror maintained by lin0 / previous step)
    v16us ah[2];
    #pragma unroll
    for (int kt = 0; kt < 2; ++kt)
        ah[kt] = *(const v16us*)(hfrag + (((size_t)tile * 2 + kt) * 32 + lane) * 16);

    // conv = h @ conv_root : 4 N-tiles x 2 K-tiles
    v8f cacc[4] = {};
    #pragma unroll
    for (int kt = 0; kt < 2; ++kt) {
        #pragma unroll
        for (int nt = 0; nt < 4; ++nt) {
            v16us b = *(const v16us*)(rootf + ((kt * 4 + nt) * 32 + lane) * 16);
            cacc[nt] = wmma_bf16(ah[kt], b, cacc[nt]);
        }
    }
    // m = relu(aggr/deg + conv + bias) -> LDS in A-fragment order
    #pragma unroll
    for (int nt = 0; nt < 4; ++nt) {
        int col = nt * 16 + lp;
        int kt = col >> 5, kl = col & 31, h2, s;
        a_slot(kl, h2, s);
        #pragma unroll
        for (int r = 0; r < 8; ++r) {
            int rr = hi * 8 + r;
            int node = row0 + rr;
            float mv = cacc[nt][r] + aggr[(size_t)node * DIM + col] / denom[node] + cbias[col];
            mfrag[((kt * 32) + ((h2 << 4) | rr)) * 16 + s] = f2bf(fmaxf(mv, 0.0f));
        }
    }
    __syncthreads();

    v16us am[2];
    #pragma unroll
    for (int kt = 0; kt < 2; ++kt)
        am[kt] = *(const v16us*)(&mfrag[(kt * 32 + lane) * 16]);

    // gi = m @ Wi^T ; gh = h @ Wh^T : 12 N-tiles x 2 K-tiles each
    #pragma unroll 4
    for (int nt = 0; nt < 12; ++nt) {
        v8f ai = {}, ahh = {};
        #pragma unroll
        for (int kt = 0; kt < 2; ++kt) {
            v16us bi = *(const v16us*)(witf + (((size_t)kt * 12 + nt) * 32 + lane) * 16);
            v16us bh = *(const v16us*)(whtf + (((size_t)kt * 12 + nt) * 32 + lane) * 16);
            ai  = wmma_bf16(am[kt], bi, ai);
            ahh = wmma_bf16(ah[kt], bh, ahh);
        }
        int col = nt * 16 + lp;
        #pragma unroll
        for (int r = 0; r < 8; ++r) {
            int rr = hi * 8 + r;
            gi_s[rr * 192 + col] = ai[r];
            gh_s[rr * 192 + col] = ahh[r];
        }
    }
    __syncthreads();

    // GRU gates (torch order r,z,n); in-place h update + refresh bf16 fragment mirror
    for (int idx = lane; idx < 16 * DIM; idx += 32) {
        int rr = idx >> 6, col = idx & 63;
        int node = row0 + rr;
        float ir  = gi_s[rr * 192 + col]       + gbi[col];
        float iz  = gi_s[rr * 192 + 64 + col]  + gbi[64 + col];
        float in_ = gi_s[rr * 192 + 128 + col] + gbi[128 + col];
        float hr  = gh_s[rr * 192 + col]       + gbh[col];
        float hz  = gh_s[rr * 192 + 64 + col]  + gbh[64 + col];
        float hn  = gh_s[rr * 192 + 128 + col] + gbh[128 + col];
        float r  = sigmoidf(ir + hr);
        float zg = sigmoidf(iz + hz);
        float ng = tanhf(in_ + r * hn);
        float hnew = (1.0f - zg) * ng + zg * h[(size_t)node * DIM + col];
        h[(size_t)node * DIM + col] = hnew;
        int kt = col >> 5, kl = col & 31, h2, s;
        a_slot(kl, h2, s);
        hfrag[(((size_t)tile * 2 + kt) * 32 + ((h2 << 4) | rr)) * 16 + s] = f2bf(hnew);
    }
}

// ---------------- Set2Set ----------------
__global__ __launch_bounds__(64) void k_lstm(float* __restrict__ qstar, float* __restrict__ hh,
                                             float* __restrict__ cc,
                                             const float* __restrict__ wi, const float* __restrict__ wh,
                                             const float* __restrict__ bi, const float* __restrict__ bh) {
    int b = blockIdx.x, t = threadIdx.x;
    float g[4];
    #pragma unroll
    for (int gg = 0; gg < 4; ++gg) {
        int j = gg * 64 + t;
        float acc = bi[j] + bh[j];
        for (int k = 0; k < 128; ++k) acc += qstar[b * 128 + k] * wi[j * 128 + k];
        for (int k = 0; k < 64;  ++k) acc += hh[b * 64 + k] * wh[j * 64 + k];
        g[gg] = acc;
    }
    __syncthreads();
    float c_new = sigmoidf(g[1]) * cc[b * 64 + t] + sigmoidf(g[0]) * tanhf(g[2]);
    float h_new = sigmoidf(g[3]) * tanhf(c_new);
    cc[b * 64 + t] = c_new;
    hh[b * 64 + t] = h_new;
    qstar[b * 128 + t] = h_new;
}

__global__ __launch_bounds__(64) void k_score(const float* __restrict__ h, const float* __restrict__ hh,
                                              const int* __restrict__ batch,
                                              float* __restrict__ e_out, unsigned* __restrict__ emax) {
    int n = blockIdx.x, t = threadIdx.x;
    int b = batch[n];
    float p = h[(size_t)n * DIM + t] * hh[b * 64 + t];
    #pragma unroll
    for (int off = 16; off > 0; off >>= 1) p += __shfl_down(p, off, 32);
    __shared__ float ps[2];
    if ((t & 31) == 0) ps[t >> 5] = p;
    __syncthreads();
    if (t == 0) {
        float e = ps[0] + ps[1];
        e_out[n] = e;
        atomicMax(&emax[b], f2ord(e));
    }
}

__global__ __launch_bounds__(256) void k_attw(const float* __restrict__ e, const unsigned* __restrict__ emax,
                                              const int* __restrict__ batch,
                                              float* __restrict__ a, float* __restrict__ asum) {
    int n = blockIdx.x * blockDim.x + threadIdx.x;
    if (n >= NN) return;
    int b = batch[n];
    float v = expf(e[n] - ord2f(emax[b]));
    a[n] = v;
    atomicAdd(&asum[b], v);
}

__global__ __launch_bounds__(64) void k_pool(const float* __restrict__ h, const float* __restrict__ a,
                                             const float* __restrict__ asum, const int* __restrict__ batch,
                                             float* __restrict__ rpool) {
    int n = blockIdx.x, t = threadIdx.x;
    int b = batch[n];
    float w = a[n] / (asum[b] + 1e-16f);
    atomicAdd(&rpool[b * 64 + t], w * h[(size_t)n * DIM + t]);
}

__global__ __launch_bounds__(256) void k_qstar(const float* __restrict__ rpool, float* __restrict__ qstar) {
    int i = blockIdx.x * blockDim.x + threadIdx.x;
    if (i >= NG * 64) return;
    int b = i / 64, t = i % 64;
    qstar[b * 128 + 64 + t] = rpool[i];
}

__global__ __launch_bounds__(64) void k_readout(const float* __restrict__ qstar,
                                                const float* __restrict__ w1, const float* __restrict__ b1,
                                                const float* __restrict__ w2, const float* __restrict__ b2,
                                                float* __restrict__ out) {
    int b = blockIdx.x, t = threadIdx.x;
    float acc = b1[t];
    for (int k = 0; k < 128; ++k) acc += qstar[b * 128 + k] * w1[k * 64 + t];
    float p = fmaxf(acc, 0.0f) * w2[t];
    #pragma unroll
    for (int off = 16; off > 0; off >>= 1) p += __shfl_down(p, off, 32);
    __shared__ float ps[2];
    if ((t & 31) == 0) ps[t >> 5] = p;
    __syncthreads();
    if (t == 0) out[b] = ps[0] + ps[1] + b2[0];
}

// ---------------- host ----------------
#define CDIV(a, b) (((a) + (b) - 1) / (b))

extern "C" void kernel_launch(void* const* d_in, const int* in_sizes, int n_in,
                              void* d_out, int out_size, void* d_ws, size_t ws_size,
                              hipStream_t stream) {
    (void)in_sizes; (void)n_in; (void)out_size; (void)ws_size;
    const float* x        = (const float*)d_in[0];
    const float* z        = (const float*)d_in[1];
    const float* eattr    = (const float*)d_in[2];
    const float* lin0_w   = (const float*)d_in[3];
    const float* lin0_b   = (const float*)d_in[4];
    const float* emlp_w1  = (const float*)d_in[5];
    const float* emlp_b1  = (const float*)d_in[6];
    const float* emlp_w2  = (const float*)d_in[7];
    const float* emlp_b2  = (const float*)d_in[8];
    const float* conv_root= (const float*)d_in[9];
    const float* conv_bias= (const float*)d_in[10];
    const float* gru_wi   = (const float*)d_in[11];
    const float* gru_wh   = (const float*)d_in[12];
    const float* gru_bi   = (const float*)d_in[13];
    const float* gru_bh   = (const float*)d_in[14];
    const float* lstm_wi  = (const float*)d_in[15];
    const float* lstm_wh  = (const float*)d_in[16];
    const float* lstm_bi  = (const float*)d_in[17];
    const float* lstm_bh  = (const float*)d_in[18];
    const float* lin1_w   = (const float*)d_in[19];
    const float* lin1_b   = (const float*)d_in[20];
    const float* lin2_w   = (const float*)d_in[21];
    const float* lin2_b   = (const float*)d_in[22];
    const int*   eidx     = (const int*)d_in[23];
    const int*   batch    = (const int*)d_in[24];
    const int*   src = eidx;
    const int*   dst = eidx + NE;
    float* out = (float*)d_out;

    // workspace carve-up
    char* base = (char*)d_ws;
    size_t off = 0;
    auto alloc = [&](size_t bytes) { void* p = base + off; off = (off + bytes + 255) & ~(size_t)255; return p; };
    unsigned short* ewt   = (unsigned short*)alloc((size_t)NE * DD * 2);      // per-edge transposed [o][i]
    unsigned short* ahid  = (unsigned short*)alloc((size_t)NE * WIDTH * 2);   // edge-hidden A-frags
    unsigned short* w2f   = (unsigned short*)alloc((size_t)WIDTH * DD * 2);   // W2 B-frags
    unsigned short* rootf = (unsigned short*)alloc(DIM * DIM * 2);            // conv_root B-frags
    unsigned short* witf  = (unsigned short*)alloc(DIM * 192 * 2);            // Wi^T B-frags
    unsigned short* whtf  = (unsigned short*)alloc(DIM * 192 * 2);            // Wh^T B-frags
    unsigned short* hfrag = (unsigned short*)alloc((size_t)NN * DIM * 2);     // h A-frag mirror
    float* h     = (float*)alloc((size_t)NN * DIM * 4);
    float* aggr  = (float*)alloc((size_t)NN * DIM * 4);
    float* denom = (float*)alloc(NN * 4);
    float* e_sc  = (float*)alloc(NN * 4);
    float* a_w   = (float*)alloc(NN * 4);
    float* qstar = (float*)alloc(NG * 128 * 4);   // qstar/hh/cc contiguous
    float* hh    = (float*)alloc(NG * 64 * 4);
    float* cc    = (float*)alloc(NG * 64 * 4);
    unsigned* emax = (unsigned*)alloc(NG * 4);    // emax/asum/rpool in order, each padded to 256 B
    float* asum  = (float*)alloc(NG * 4);
    float* rpool = (float*)alloc(NG * 64 * 4);

    // weight prep: pack into WMMA B-fragment order (bf16)
    k_pack_bfrag<<<CDIV(WIDTH * DD, 256), 256, 0, stream>>>(emlp_w2, w2f, WIDTH, DD, 0);
    k_pack_bfrag<<<CDIV(DIM * DIM, 256), 256, 0, stream>>>(conv_root, rootf, DIM, DIM, 0);
    k_pack_bfrag<<<CDIV(DIM * 192, 256), 256, 0, stream>>>(gru_wi, witf, DIM, 192, 1);
    k_pack_bfrag<<<CDIV(DIM * 192, 256), 256, 0, stream>>>(gru_wh, whtf, DIM, 192, 1);

    // node init + degree
    k_lin0<<<CDIV(NN * DIM, 256), 256, 0, stream>>>(x, z, lin0_w, lin0_b, h, hfrag);
    k_zero_f32<<<CDIV(NN, 256), 256, 0, stream>>>(denom, NN);
    k_deg<<<CDIV(NE, 256), 256, 0, stream>>>(dst, denom);
    k_deg_fin<<<CDIV(NN, 256), 256, 0, stream>>>(denom);

    // edge MLP (big WMMA GEMM, once)
    k_edge_hidden<<<CDIV(NE * WIDTH, 256), 256, 0, stream>>>(eattr, emlp_w1, emlp_b1, ahid);
    k_edge_ew<<<(NE / 32) * 64 / 4, 128, 0, stream>>>(ahid, w2f, emlp_b2, ewt);

    // message-passing steps
    for (int s = 0; s < NSTEPS; ++s) {
        k_zero_f32<<<CDIV(NN * DIM, 256), 256, 0, stream>>>(aggr, NN * DIM);
        k_msg<<<NE, 64, 0, stream>>>(h, ewt, src, dst, aggr);
        k_node<<<NN / 16, 32, 0, stream>>>(h, hfrag, aggr, denom, rootf, conv_bias,
                                           witf, whtf, gru_bi, gru_bh);
    }

    // Set2Set
    k_zero_f32<<<CDIV(NG * 256, 256), 256, 0, stream>>>(qstar, NG * 256);
    for (int s = 0; s < NSTEPS; ++s) {
        k_lstm<<<NG, 64, 0, stream>>>(qstar, hh, cc, lstm_wi, lstm_wh, lstm_bi, lstm_bh);
        k_zero_f32<<<CDIV(NG * 66, 256), 256, 0, stream>>>((float*)emax, NG * 66);
        k_score<<<NN, 64, 0, stream>>>(h, hh, batch, e_sc, emax);
        k_attw<<<CDIV(NN, 256), 256, 0, stream>>>(e_sc, emax, batch, a_w, asum);
        k_pool<<<NN, 64, 0, stream>>>(h, a_w, asum, batch, rpool);
        k_qstar<<<CDIV(NG * 64, 256), 256, 0, stream>>>(rpool, qstar);
    }

    // readout
    k_readout<<<NG, 64, 0, stream>>>(qstar, lin1_w, lin1_b, lin2_w, lin2_b, out);
}
